// QuantumDiffusionDreamer_67525475827879
// MI455X (gfx1250) — matrix operand order repaired and checked by reading, not compiled, not run on hardware
//
#include <hip/hip_runtime.h>
#include <hip/hip_bf16.h>
#include <stdint.h>

// ---------------------------------------------------------------------------
// Types for CDNA5 WMMA (wave32, 16x16x32 bf16 -> f32)
// ---------------------------------------------------------------------------
typedef __attribute__((ext_vector_type(16))) __bf16 bfx16;
typedef __attribute__((ext_vector_type(8)))  float  fx8;

union Frag16 { uint32_t u[8]; bfx16 v; };

#define BM 128
#define BN 128
#define BK 32
#define GEMM_THREADS 256   // 8 wave32 waves: 4 along M x 2 along N, wave tile 32x64

// ---------------------------------------------------------------------------
// Helpers
// ---------------------------------------------------------------------------
__device__ __forceinline__ unsigned short f32_to_bf16_rne(float f) {
  uint32_t u = __float_as_uint(f);
  uint32_t r = 0x7FFFu + ((u >> 16) & 1u);
  return (unsigned short)((u + r) >> 16);
}

__device__ __forceinline__ uint64_t mix64(uint64_t h) {
  h ^= h >> 33; h *= 0xFF51AFD7ED558CCDull;
  h ^= h >> 33; h *= 0xC4CEB9FE1A85EC53ull;
  h ^= h >> 33; return h;
}

__device__ __forceinline__ float hash_normal(uint32_t idx, uint32_t seed) {
  uint64_t h = mix64((((uint64_t)seed) << 32) | (uint64_t)idx) ^ 0x9E3779B97F4A7C15ull;
  h = mix64(h);
  uint32_t u1 = (uint32_t)h, u2 = (uint32_t)(h >> 32);
  float f1 = ((float)u1 + 1.0f) * 2.3283064365386963e-10f;   // (0,1]
  float f2 = (float)u2 * 2.3283064365386963e-10f;            // [0,1)
  return sqrtf(-2.0f * __logf(f1)) * __cosf(6.28318530717958647f * f2);
}

__device__ __forceinline__ float hash_uniform(uint32_t seed) {
  uint64_t h = mix64(0x0123456789ABCDEFull ^ (uint64_t)seed);
  return (float)((uint32_t)h) * 2.3283064365386963e-10f;
}

// Extract wave-relative LDS byte offset from a __shared__ pointer (low 32 bits
// of the generic address are the LDS offset on CDNA5).
__device__ __forceinline__ uint32_t lds_off(const void* p) {
  return (uint32_t)(uintptr_t)p;
}

// Async global -> LDS copy of 16 bytes (tracked by ASYNCcnt).
__device__ __forceinline__ void async_ld_b128(uint32_t lds_addr, const void* gptr) {
  asm volatile("global_load_async_to_lds_b128 %0, %1, off"
               :: "v"(lds_addr), "v"((uint64_t)(uintptr_t)gptr)
               : "memory");
}

__device__ __forceinline__ void wait_async0() {
  asm volatile("s_wait_asynccnt 0x0" ::: "memory");
}

// ---------------------------------------------------------------------------
// Tiled bf16 WMMA GEMM:  D = act(A[MxK] * BT^T + bias),  BT is [N x K] (B^T)
//   - double-buffered LDS, async global->LDS staging, 1 barrier / K-iter
//   - outBf (optional): bf16 output, leading dim ldBf (feeds next GEMM)
//   - outF  (optional): f32 output = scale * D   (or += if accF), ld ldF
//   - scale comes from a device pointer (quantum amplitudes live on device)
// ---------------------------------------------------------------------------
__global__ __launch_bounds__(GEMM_THREADS)
void qdd_gemm_bf16(const unsigned short* __restrict__ A,
                   const unsigned short* __restrict__ BT,
                   const float* __restrict__ bias,
                   unsigned short* __restrict__ outBf,
                   float* __restrict__ outF,
                   const float* __restrict__ scalePtr,
                   int M, int N, int K, int ldBf, int ldF,
                   int act, int accF) {
  __shared__ unsigned short As[2][BM][BK + 8];   // [m][k]
  __shared__ unsigned short Bs[2][BN][BK + 8];   // [n][k]  (B already transposed)

  const int tid    = threadIdx.x;
  const int lane   = tid & 31;
  const int wave   = tid >> 5;
  const int waveM  = wave & 3;
  const int waveN  = wave >> 2;
  const int blockM = blockIdx.x * BM;
  const int blockN = blockIdx.y * BN;

  fx8 acc[2][4];
#pragma unroll
  for (int r = 0; r < 2; ++r)
#pragma unroll
    for (int c = 0; c < 4; ++c) {
      fx8 z = {0.f, 0.f, 0.f, 0.f, 0.f, 0.f, 0.f, 0.f};
      acc[r][c] = z;
    }

  // staging: each thread moves 32B of A-tile and 32B of B-tile per K-step
  const int sRow = tid >> 1;             // 0..127 (row of A tile / row of BT tile)
  const int sCol = (tid & 1) << 4;       // 0 or 16 (halves)

  // fragment lane decode (per CDNA5 ISA 16-bit A/B layouts)
  const int mLane = lane & 15;
  const int aKSel = (lane >> 4) << 3;    // lanes 16-31 offset K by 8
  const int bKSel = (lane >> 4) << 3;    // lanes 16-31 offset K by 8

  const unsigned short* aSrcBase = A  + (size_t)(blockM + sRow) * K + sCol;
  const unsigned short* bSrcBase = BT + (size_t)(blockN + sRow) * K + sCol;

  auto stage = [&](int buf, int k0) {
    const unsigned short* ag = aSrcBase + k0;
    const unsigned short* bg = bSrcBase + k0;
    uint32_t la = lds_off(&As[buf][sRow][sCol]);
    uint32_t lb = lds_off(&Bs[buf][sRow][sCol]);
    async_ld_b128(la,      ag);
    async_ld_b128(la + 16, ag + 8);
    async_ld_b128(lb,      bg);
    async_ld_b128(lb + 16, bg + 8);
  };

  // prologue: stage first tile
  stage(0, 0);
  wait_async0();
  __syncthreads();

  int cur = 0;
  for (int k0 = 0; k0 < K; k0 += BK) {
    const int nxt = k0 + BK;
    if (nxt < K) {
      stage(cur ^ 1, nxt);               // async into ping-pong buffer
      if (nxt + BK < K) {                // L2 prefetch two tiles ahead
        __builtin_prefetch(aSrcBase + nxt + BK, 0, 1);
        __builtin_prefetch(bSrcBase + nxt + BK, 0, 1);
      }
    }

    // ---- fragment loads from LDS (current buffer) ----
    Frag16 af[2];
#pragma unroll
    for (int r = 0; r < 2; ++r) {
      const int row = waveM * 32 + r * 16 + mLane;
#pragma unroll
      for (int p = 0; p < 8; ++p) {
        const int kk = ((p >= 4) ? 16 : 0) + aKSel + ((p & 3) << 1);
        af[r].u[p] = *reinterpret_cast<const uint32_t*>(&As[cur][row][kk]);
      }
    }
    Frag16 bfm[4];
#pragma unroll
    for (int c = 0; c < 4; ++c) {
      const int col = waveN * 64 + c * 16 + mLane;
#pragma unroll
      for (int p = 0; p < 8; ++p) {
        const int kk = ((p >= 4) ? 16 : 0) + bKSel + ((p & 3) << 1);
        bfm[c].u[p] = *reinterpret_cast<const uint32_t*>(&Bs[cur][col][kk]);
      }
    }

    // ---- 8 WMMAs per K-iteration ----
#pragma unroll
    for (int r = 0; r < 2; ++r)
#pragma unroll
      for (int c = 0; c < 4; ++c)
        acc[r][c] = __builtin_amdgcn_wmma_f32_16x16x32_bf16(
            false, af[r].v, false, bfm[c].v, (short)0, acc[r][c], false, false);

    wait_async0();                       // next tile fully in LDS
    __syncthreads();                     // single barrier per iteration
    cur ^= 1;
  }

  // ---- epilogue: bias + activation + dual-format store ----
  const float scl  = scalePtr ? scalePtr[0] : 1.0f;
  const int   mOff = (lane >> 4) << 3;   // C/D layout: lanes 16-31 hold M+8
  const int   nOff = lane & 15;
#pragma unroll
  for (int r = 0; r < 2; ++r) {
#pragma unroll
    for (int c = 0; c < 4; ++c) {
      const int   colG = blockN + waveN * 64 + c * 16 + nOff;
      const float bs   = bias ? bias[colG] : 0.0f;
#pragma unroll
      for (int v = 0; v < 8; ++v) {
        const int rowG = blockM + waveM * 32 + r * 16 + mOff + v;
        float val = acc[r][c][v] + bs;
        if (act == 1)      val = fmaxf(val, 0.0f);                                  // relu
        else if (act == 2) val = tanhf(val);                                        // tanh
        else if (act == 3) val = 0.5f * val * (1.0f + erff(val * 0.70710678118f));  // exact gelu
        if (outBf)
          outBf[(size_t)rowG * ldBf + colG] = f32_to_bf16_rne(val);
        if (outF) {
          const size_t o  = (size_t)rowG * ldF + colG;
          const float  ov = scl * val;
          outF[o] = accF ? (outF[o] + ov) : ov;
        }
      }
    }
  }
}

// ---------------------------------------------------------------------------
// Elementwise kernels
// ---------------------------------------------------------------------------

// f32 [K x N] -> bf16 transposed [N x K] (one-time weight prep)
__global__ void qdd_cvt_bf16_t(const float* __restrict__ in,
                               unsigned short* __restrict__ out,
                               int K, int N) {
  int i = blockIdx.x * blockDim.x + threadIdx.x;
  if (i >= K * N) return;
  int k = i / N, n = i - k * N;
  out[(size_t)n * K + k] = f32_to_bf16_rne(in[i]);
}

__global__ void qdd_scalars(const float* __restrict__ aAmp,
                            const float* __restrict__ bAmp,
                            float* __restrict__ scal) {
  if (blockIdx.x == 0 && threadIdx.x == 0) {
    float a2 = aAmp[0] * aAmp[0], b2 = bAmp[0] * bAmp[0];
    float p0 = a2 / (a2 + b2), p1 = b2 / (a2 + b2);
    scal[0] = sqrtf(p0);
    scal[1] = sqrtf(p1);
  }
}

__global__ void qdd_init_x(float* __restrict__ x, unsigned short* __restrict__ xb, int n) {
  int i = blockIdx.x * blockDim.x + threadIdx.x;
  if (i >= n) return;
  float v = hash_normal((uint32_t)i, 10000u);
  x[i]  = v;
  xb[i] = f32_to_bf16_rne(v);
}

__global__ void qdd_update_x(float* __restrict__ x, unsigned short* __restrict__ xb,
                             const float* __restrict__ partial,
                             const float* __restrict__ interf,
                             const float* __restrict__ aAmp,
                             const float* __restrict__ bAmp,
                             const float* __restrict__ pCos,
                             int stepIdx, int steps, int n) {
  int i = blockIdx.x * blockDim.x + threadIdx.x;
  if (i >= n) return;
  float a2 = aAmp[0] * aAmp[0], b2 = bAmp[0] * bAmp[0];
  float p0 = a2 / (a2 + b2), p1 = b2 / (a2 + b2);
  float coh = __expf(-0.1f * (float)stepIdx);          // decay^stepIdx
  float qi  = 2.0f * sqrtf(p0 * p1) * pCos[0] * coh;

  int   t       = steps - 1 - stepIdx;
  float sched_t = 1e-4f + (0.02f - 1e-4f) * ((float)t / 99.0f);
  float alpha   = fmaxf(1.0f - sched_t, 1e-8f);
  float alpha_p = 1.0f;
  if (t > 0) {
    float sp = 1e-4f + (0.02f - 1e-4f) * ((float)(t - 1) / 99.0f);
    alpha_p  = 1.0f - sp;
  }
  float s1ma = sqrtf(fmaxf(1.0f - alpha, 1e-8f));
  float den  = partial[i] + qi * interf[i];
  float xn   = (x[i] - (1.0f - alpha) / s1ma * den) * rsqrtf(alpha);
  if (t > 0) {
    float sigma = sqrtf(fmaxf((1.0f - alpha_p) / (1.0f - alpha) * (1.0f - alpha / alpha_p), 0.0f));
    xn += (sigma + 0.1f * fabsf(qi)) * hash_normal((uint32_t)i, (uint32_t)t);
  }
  x[i]  = xn;
  xb[i] = f32_to_bf16_rne(xn);
}

__global__ void qdd_finalize(const float* __restrict__ x,
                             const float* __restrict__ real,
                             const float* __restrict__ aAmp,
                             const float* __restrict__ bAmp,
                             float* __restrict__ out, int n) {
  int i = blockIdx.x * blockDim.x + threadIdx.x;
  if (i >= n) return;
  float a2 = aAmp[0] * aAmp[0], b2 = bAmp[0] * bAmp[0];
  float p0 = a2 / (a2 + b2);
  float u  = hash_uniform(99999u);
  out[i] = (u < p0) ? (0.7f * x[i] + 0.3f * real[i]) : x[i];
}

// ---------------------------------------------------------------------------
// Launch
// ---------------------------------------------------------------------------
extern "C" void kernel_launch(void* const* d_in, const int* in_sizes, int n_in,
                              void* d_out, int out_size, void* d_ws, size_t ws_size,
                              hipStream_t stream) {
  (void)in_sizes; (void)n_in; (void)out_size; (void)ws_size;

  const float* real = (const float*)d_in[0];
  const float* Wq   = (const float*)d_in[2];
  const float* bq   = (const float*)d_in[3];
  const float* W01  = (const float*)d_in[4];
  const float* b01  = (const float*)d_in[5];
  const float* W02  = (const float*)d_in[6];
  const float* b02  = (const float*)d_in[7];
  const float* W11  = (const float*)d_in[8];
  const float* b11  = (const float*)d_in[9];
  const float* W12  = (const float*)d_in[10];
  const float* b12  = (const float*)d_in[11];
  const float* Wi1  = (const float*)d_in[12];
  const float* bi1  = (const float*)d_in[13];
  const float* Wi2  = (const float*)d_in[14];
  const float* bi2  = (const float*)d_in[15];
  const float* aAmp = (const float*)d_in[16];
  const float* bAmp = (const float*)d_in[17];
  const float* pCos = (const float*)d_in[18];

  const int NR = 8192, F = 512, H = 1024, STEPS_C = 50;
  const int nxf = NR * F;

  char* ws = (char*)d_ws;
  size_t off = 0;
  auto alloc = [&](size_t bytes) -> char* {
    char* p = ws + off;
    off += (bytes + 255) & ~(size_t)255;
    return p;
  };
  float*          x      = (float*)         alloc((size_t)NR * F * 4);
  unsigned short* xb     = (unsigned short*)alloc((size_t)NR * F * 2);
  unsigned short* qf     = (unsigned short*)alloc((size_t)NR * H * 2);
  unsigned short* hbuf   = (unsigned short*)alloc((size_t)NR * H * 2);
  unsigned short* dcat   = (unsigned short*)alloc((size_t)NR * H * 2);  // [d0|d1]
  unsigned short* gbuf   = (unsigned short*)alloc((size_t)NR * H * 2);
  float*          part   = (float*)         alloc((size_t)NR * F * 4);  // sp0*d0+sp1*d1
  float*          interf = (float*)         alloc((size_t)NR * F * 4);
  unsigned short* WqT    = (unsigned short*)alloc((size_t)F * H * 2);   // [H x F]
  unsigned short* W01T   = (unsigned short*)alloc((size_t)H * H * 2);
  unsigned short* W02T   = (unsigned short*)alloc((size_t)H * F * 2);   // [F x H]
  unsigned short* W11T   = (unsigned short*)alloc((size_t)H * H * 2);
  unsigned short* W12T   = (unsigned short*)alloc((size_t)H * F * 2);
  unsigned short* Wi1T   = (unsigned short*)alloc((size_t)H * H * 2);   // 2F == H
  unsigned short* Wi2T   = (unsigned short*)alloc((size_t)H * F * 2);
  float*          scal   = (float*)         alloc(256);

  // ---- one-time per launch: weight conversion (transposed) + scalars ----
  auto cvtT = [&](const float* src, unsigned short* dst, int K, int N) {
    int n = K * N;
    qdd_cvt_bf16_t<<<(n + 255) / 256, 256, 0, stream>>>(src, dst, K, N);
  };
  cvtT(Wq,  WqT,  F, H);   // Wq  [F x H] -> WqT  [H x F]
  cvtT(W01, W01T, H, H);
  cvtT(W02, W02T, H, F);   // W02 [H x F] -> W02T [F x H]
  cvtT(W11, W11T, H, H);
  cvtT(W12, W12T, H, F);
  cvtT(Wi1, Wi1T, H, H);
  cvtT(Wi2, Wi2T, H, F);
  qdd_scalars<<<1, 32, 0, stream>>>(aAmp, bAmp, scal);
  qdd_init_x<<<(nxf + 255) / 256, 256, 0, stream>>>(x, xb, nxf);

  const dim3 blk(GEMM_THREADS);
  const dim3 gH(NR / BM, H / BN);   // N=1024 outputs
  const dim3 gF(NR / BM, F / BN);   // N=512  outputs

  for (int s = 0; s < STEPS_C; ++s) {
    // qf = relu(x @ Wq + bq)                         [NR,H]
    qdd_gemm_bf16<<<gH, blk, 0, stream>>>(xb, WqT, bq, qf, nullptr, nullptr,
                                          NR, H, F, H, 0, /*relu*/1, 0);
    // h = relu(qf @ W01 + b01)                       [NR,H]
    qdd_gemm_bf16<<<gH, blk, 0, stream>>>(qf, W01T, b01, hbuf, nullptr, nullptr,
                                          NR, H, H, H, 0, /*relu*/1, 0);
    // d0 = h @ W02 + b02  -> dcat[:, :512] (bf16), part = sp0*d0 (f32)
    qdd_gemm_bf16<<<gF, blk, 0, stream>>>(hbuf, W02T, b02, dcat, part, scal,
                                          NR, F, H, H, F, /*none*/0, 0);
    // h = tanh(qf @ W11 + b11)
    qdd_gemm_bf16<<<gH, blk, 0, stream>>>(qf, W11T, b11, hbuf, nullptr, nullptr,
                                          NR, H, H, H, 0, /*tanh*/2, 0);
    // d1 = h @ W12 + b12  -> dcat[:, 512:], part += sp1*d1
    qdd_gemm_bf16<<<gF, blk, 0, stream>>>(hbuf, W12T, b12, dcat + F, part, scal + 1,
                                          NR, F, H, H, F, /*none*/0, 1);
    // g = gelu(dcat @ Wi1 + bi1)                     [NR,H]
    qdd_gemm_bf16<<<gH, blk, 0, stream>>>(dcat, Wi1T, bi1, gbuf, nullptr, nullptr,
                                          NR, H, H, H, 0, /*gelu*/3, 0);
    // interf = g @ Wi2 + bi2                         [NR,F] f32
    qdd_gemm_bf16<<<gF, blk, 0, stream>>>(gbuf, Wi2T, bi2, nullptr, interf, nullptr,
                                          NR, F, H, 0, F, /*none*/0, 0);
    // DDPM update + noise + bf16 recast of x
    qdd_update_x<<<(nxf + 255) / 256, 256, 0, stream>>>(x, xb, part, interf,
                                                        aAmp, bAmp, pCos,
                                                        s, STEPS_C, nxf);
  }

  qdd_finalize<<<(nxf + 255) / 256, 256, 0, stream>>>(x, real, aAmp, bAmp,
                                                      (float*)d_out, nxf);
}
